// Dim6RotStructureHead_32083405701178
// MI455X (gfx1250) — matrix operand order repaired and checked by reading, not compiled, not run on hardware
//
#include <hip/hip_runtime.h>
#include <math.h>

// ---------------------------------------------------------------------------
// Dim6RotStructureHead for MI455X (gfx1250, wave32, WMMA + TDM)
// B=16, L=4096, D=512  ->  65536 rows
// ---------------------------------------------------------------------------

#define NROWS   65536
#define DDIM    512
#define TRANS_SCALE 10.0f
#define LN_EPS  1e-5f

typedef __attribute__((ext_vector_type(8)))  float        v8f;
typedef __attribute__((ext_vector_type(16))) __bf16       v16bf;
typedef __attribute__((ext_vector_type(4)))  unsigned int u32x4;
typedef __attribute__((ext_vector_type(8)))  int          i32x8;
typedef __attribute__((ext_vector_type(4)))  int          i32x4;

union BFrag { uint4 u[2]; v16bf v; };
union BF16Bits { __bf16 h; unsigned short s; };

#if __has_builtin(__builtin_amdgcn_tensor_load_to_lds) && __has_builtin(__builtin_amdgcn_s_wait_tensorcnt)
#define HAS_TDM 1
#else
#define HAS_TDM 0
#endif

__device__ __forceinline__ float fast_tanh(float x) {
#if __has_builtin(__builtin_amdgcn_tanh_f32)
    return __builtin_amdgcn_tanh_f32(x);
#else
    return tanhf(x);
#endif
}
__device__ __forceinline__ float gelu_f(float v) {
    // tanh-form GELU (uses gfx1250 hardware v_tanh_f32)
    float u = 0.7978845608028654f * v * (1.0f + 0.044715f * v * v);
    return 0.5f * v * (1.0f + fast_tanh(u));
}

// ---------------------------------------------------------------------------
// Pre-pass 1: W1 (512x512 f32 row-major) -> bf16 WMMA-B fragment tiles.
// Tile (kt,nt): 32(K) x 16(N) bf16, 1 KB. K-panel kt = 32 tiles = 32 KB,
// contiguous, so the TDM can stream one panel per K-step.
// Within a tile: half h (m<8 / m>=8) at h*512B, lane's 16B at lane*16B.
//   halfword m (0..15): K = (m>=8?16:0)+(lane>=16?8:0)+m%8 , N = lane%16
// ---------------------------------------------------------------------------
__global__ void convert_w1_kernel(const float* __restrict__ W1,
                                  unsigned short* __restrict__ Wt) {
    int idx  = blockIdx.x * 256 + threadIdx.x;      // 0 .. 262143
    int e    = idx & 7;
    int lane = (idx >> 3) & 31;
    int h    = (idx >> 8) & 1;
    int tile = idx >> 9;
    int kt   = tile >> 5;
    int nt   = tile & 31;
    int m    = h * 8 + e;
    int K    = h * 16 + ((lane >= 16) ? 8 : 0) + e;
    int N    = lane & 15;
    float v  = W1[(kt * 32 + K) * DDIM + nt * 16 + N];
    BF16Bits b; b.h = (__bf16)v;
    Wt[idx] = b.s;                                  // coalesced
}

// ---------------------------------------------------------------------------
// Pre-pass 2: Cw[q] = sum_c lnw[c]*Wp[c,q],  Cb[q] = sum_c lnb[c]*Wp[c,q]
// ---------------------------------------------------------------------------
__global__ void consts_kernel(const float* __restrict__ lnw,
                              const float* __restrict__ lnb,
                              const float* __restrict__ Wp,
                              float* __restrict__ cst) {
    int q = threadIdx.x >> 5;
    int lane = threadIdx.x & 31;
    if (q < 9) {
        float sw = 0.0f, sb = 0.0f;
        for (int c = lane; c < DDIM; c += 32) {
            float wp = Wp[c * 9 + q];
            sw += lnw[c] * wp;
            sb += lnb[c] * wp;
        }
        sw += __shfl_xor(sw, 16); sb += __shfl_xor(sb, 16);
        sw += __shfl_xor(sw, 8);  sb += __shfl_xor(sb, 8);
        sw += __shfl_xor(sw, 4);  sb += __shfl_xor(sb, 4);
        sw += __shfl_xor(sw, 2);  sb += __shfl_xor(sb, 2);
        sw += __shfl_xor(sw, 1);  sb += __shfl_xor(sb, 1);
        if (lane == 0) { cst[q] = sw; cst[9 + q] = sb; }
    }
}

#if HAS_TDM
// TDM: DMA one contiguous 32 KB B-panel from global into LDS.
// D# per CDNA5 ISA: 1-D tile, data_size=8B, 4096 elems.
__device__ __forceinline__ void tdm_load_panel(const unsigned short* gsrc,
                                               void* ldsdst) {
    unsigned long long ga = (unsigned long long)(uintptr_t)gsrc;
    unsigned int lds = (unsigned int)(uintptr_t)ldsdst;   // low 32 = LDS offset
    u32x4 g0;
    g0[0] = 1u;                                            // count=1 (valid)
    g0[1] = lds;                                           // lds_addr
    g0[2] = (unsigned int)ga;                              // global_addr lo
    g0[3] = (unsigned int)((ga >> 32) & 0x01FFFFFFu)       // global_addr hi
          | 0x80000000u;                                   // type=2 (image)
    i32x8 g1;
    g1[0] = (int)(3u << 16);        // wg_mask=0, data_size=3 (8B)
    g1[1] = (int)(4096u << 16);     // tensor_dim0[15:0]=4096
    g1[2] = (int)(1u << 16);        // tensor_dim0 hi=0 | tensor_dim1=1
    g1[3] = (int)(4096u << 16);     // tensor_dim1 hi=0 | tile_dim0=4096
    g1[4] = 0;                      // tile_dim1=0, tile_dim2=0 (unused)
    g1[5] = 4096;                   // tensor_dim0_stride lo32
    g1[6] = 0;
    g1[7] = 0;
    i32x4 z4 = (i32x4)0;
    i32x8 z8 = (i32x8)0;
    __builtin_amdgcn_tensor_load_to_lds(g0, g1, z4, z4, z8, 0);
}
#endif

// ---------------------------------------------------------------------------
// Main fused kernel.
// 1024 threads = 32 waves; block = 128 rows x 512 cols.
// wave (waveRow 0..3, waveCol 0..7) owns 32 rows x 64 cols
//   -> acc[2][4] 16x16 f32 fragments (64 VGPRs)  [no spills, B dbl-buffer]
// B panels (32 KB per K-step) double-buffered in LDS via TDM.
// ---------------------------------------------------------------------------
__global__ __launch_bounds__(1024)
void structure_head_kernel(const float* __restrict__ x,
                           const unsigned short* __restrict__ W1bf,
                           const float* __restrict__ b1,
                           const float* __restrict__ lnw,
                           const float* __restrict__ Wp,
                           const float* __restrict__ bp,
                           const float* __restrict__ cst,
                           const float* __restrict__ affine,
                           const unsigned char* __restrict__ amask,
                           float* __restrict__ out_aff,
                           float* __restrict__ out_xyz) {
    __shared__ unsigned short bpanel[2][16384];   // 2 x 32 KB B panels
    __shared__ float wqsm[DDIM * 9];              // lnw[c]*Wp[c][q]
    __shared__ float psum[128][8][12];            // per-row partials (s1,s2,d0..8)

    const int tid     = threadIdx.x;
    const int lane    = tid & 31;
    const int wave    = tid >> 5;
    const int waveRow = wave >> 3;                // 0..3  (32-row groups)
    const int waveCol = wave & 7;                 // 0..7  (64-col groups)
    const int halfSel = lane >> 4;
    const int l15     = lane & 15;
    const int rowBlock = blockIdx.x * 128;

    // ---- stage lnw*Wp into LDS (one column per thread) --------------------
    if (tid < DDIM) {
        float lw = lnw[tid];
#pragma unroll
        for (int q = 0; q < 9; ++q) wqsm[tid * 9 + q] = lw * Wp[tid * 9 + q];
    }

#if HAS_TDM
    if (wave == 0) {
        tdm_load_panel(W1bf, &bpanel[0][0]);                  // panel kt=0
        __builtin_amdgcn_s_wait_tensorcnt(0);
    }
    __syncthreads();
#endif

    // ---- GEMM1: h = gelu(x @ W1 + b1) -------------------------------------
    const float* xrow0 = x + (size_t)(rowBlock + waveRow * 32 + l15) * DDIM;
    const float* xrow1 = xrow0 + (size_t)16 * DDIM;

    v8f acc[2][4];
#pragma unroll
    for (int m = 0; m < 2; ++m)
#pragma unroll
        for (int t = 0; t < 4; ++t) acc[m][t] = (v8f)0.0f;

    for (int kt = 0; kt < 16; ++kt) {
#if !HAS_TDM
        // fallback: cooperative global->LDS copy of panel kt
        __syncthreads();
        {
            const uint4* src = (const uint4*)(W1bf + (size_t)kt * 16384);
            uint4* dst = (uint4*)&bpanel[kt & 1][0];
            for (int i = tid; i < 2048; i += 1024) dst[i] = src[i];
        }
        __syncthreads();
#else
        if (kt < 15 && wave == 0)
            tdm_load_panel(W1bf + (size_t)(kt + 1) * 16384,
                           &bpanel[(kt + 1) & 1][0]);
#endif
        // preload all 4 B fragments of this K-step (one ds clause, one wait)
        const uint4* bbuf = (const uint4*)&bpanel[kt & 1][0]
                          + (waveCol * 4) * 64 + lane;
        BFrag f[4];
#pragma unroll
        for (int t = 0; t < 4; ++t) {
            f[t].u[0] = bbuf[t * 64];
            f[t].u[1] = bbuf[t * 64 + 32];
        }

        // A fragments (two 16-row groups), f32 -> bf16
        const int off = kt * 32 + halfSel * 8;
        const float4 a0a = *(const float4*)(xrow0 + off);
        const float4 a0b = *(const float4*)(xrow0 + off + 4);
        const float4 a0c = *(const float4*)(xrow0 + off + 16);
        const float4 a0d = *(const float4*)(xrow0 + off + 20);
        const float4 a1a = *(const float4*)(xrow1 + off);
        const float4 a1b = *(const float4*)(xrow1 + off + 4);
        const float4 a1c = *(const float4*)(xrow1 + off + 16);
        const float4 a1d = *(const float4*)(xrow1 + off + 20);
        v16bf a0, a1;
        a0[0]=(__bf16)a0a.x; a0[1]=(__bf16)a0a.y; a0[2]=(__bf16)a0a.z; a0[3]=(__bf16)a0a.w;
        a0[4]=(__bf16)a0b.x; a0[5]=(__bf16)a0b.y; a0[6]=(__bf16)a0b.z; a0[7]=(__bf16)a0b.w;
        a0[8]=(__bf16)a0c.x; a0[9]=(__bf16)a0c.y; a0[10]=(__bf16)a0c.z; a0[11]=(__bf16)a0c.w;
        a0[12]=(__bf16)a0d.x; a0[13]=(__bf16)a0d.y; a0[14]=(__bf16)a0d.z; a0[15]=(__bf16)a0d.w;
        a1[0]=(__bf16)a1a.x; a1[1]=(__bf16)a1a.y; a1[2]=(__bf16)a1a.z; a1[3]=(__bf16)a1a.w;
        a1[4]=(__bf16)a1b.x; a1[5]=(__bf16)a1b.y; a1[6]=(__bf16)a1b.z; a1[7]=(__bf16)a1b.w;
        a1[8]=(__bf16)a1c.x; a1[9]=(__bf16)a1c.y; a1[10]=(__bf16)a1c.z; a1[11]=(__bf16)a1c.w;
        a1[12]=(__bf16)a1d.x; a1[13]=(__bf16)a1d.y; a1[14]=(__bf16)a1d.z; a1[15]=(__bf16)a1d.w;

#pragma unroll
        for (int t = 0; t < 4; ++t) {
            acc[0][t] = __builtin_amdgcn_wmma_f32_16x16x32_bf16(
                false, a0, false, f[t].v, (short)0, acc[0][t], false, false);
            acc[1][t] = __builtin_amdgcn_wmma_f32_16x16x32_bf16(
                false, a1, false, f[t].v, (short)0, acc[1][t], false, false);
        }
#if HAS_TDM
        if (kt < 15 && wave == 0) __builtin_amdgcn_s_wait_tensorcnt(0);
#endif
        __syncthreads();
    }

    // ---- epilogue: bias + GELU in-place on accumulators -------------------
    float bias[4];
#pragma unroll
    for (int t = 0; t < 4; ++t) bias[t] = b1[waveCol * 64 + t * 16 + l15];
#pragma unroll
    for (int m = 0; m < 2; ++m)
#pragma unroll
        for (int t = 0; t < 4; ++t)
#pragma unroll
            for (int r = 0; r < 8; ++r)
                acc[m][t][r] = gelu_f(acc[m][t][r] + bias[t]);

    const int rbaseW = waveRow * 32;

    // s1 / s2 partials per row (reduce within 16-lane halves)
#pragma unroll
    for (int m = 0; m < 2; ++m)
#pragma unroll
        for (int r = 0; r < 8; ++r) {
            float s1 = 0.0f, s2 = 0.0f;
#pragma unroll
            for (int t = 0; t < 4; ++t) {
                float g = acc[m][t][r];
                s1 += g; s2 += g * g;
            }
            s1 += __shfl_xor(s1, 1); s2 += __shfl_xor(s2, 1);
            s1 += __shfl_xor(s1, 2); s2 += __shfl_xor(s2, 2);
            s1 += __shfl_xor(s1, 4); s2 += __shfl_xor(s2, 4);
            s1 += __shfl_xor(s1, 8); s2 += __shfl_xor(s2, 8);
            if (l15 == 0) {
                int row = rbaseW + m * 16 + halfSel * 8 + r;
                psum[row][waveCol][0] = s1;
                psum[row][waveCol][1] = s2;
            }
        }

    // 9-dot partials: d[q] = sum_c h_c * (lnw*Wp)[c][q]
#pragma unroll
    for (int q = 0; q < 9; ++q) {
        float wqt[4];
#pragma unroll
        for (int t = 0; t < 4; ++t)
            wqt[t] = wqsm[(waveCol * 64 + t * 16 + l15) * 9 + q];
#pragma unroll
        for (int m = 0; m < 2; ++m)
#pragma unroll
            for (int r = 0; r < 8; ++r) {
                float d = 0.0f;
#pragma unroll
                for (int t = 0; t < 4; ++t) d += acc[m][t][r] * wqt[t];
                d += __shfl_xor(d, 1);
                d += __shfl_xor(d, 2);
                d += __shfl_xor(d, 4);
                d += __shfl_xor(d, 8);
                if (l15 == 0)
                    psum[rbaseW + m * 16 + halfSel * 8 + r][waveCol][2 + q] = d;
            }
    }
    __syncthreads();

    // ---- per-row combine: LN stats + projection + geometry ----------------
    if (tid < 128) {
        const int row = tid;
        const int g = rowBlock + row;
        float S1 = 0.0f, S2 = 0.0f, D[9];
#pragma unroll
        for (int q = 0; q < 9; ++q) D[q] = 0.0f;
#pragma unroll
        for (int wc = 0; wc < 8; ++wc) {
            S1 += psum[row][wc][0];
            S2 += psum[row][wc][1];
#pragma unroll
            for (int q = 0; q < 9; ++q) D[q] += psum[row][wc][2 + q];
        }
        const float mu   = S1 * (1.0f / 512.0f);
        const float var  = S2 * (1.0f / 512.0f) - mu * mu;
        const float rstd = rsqrtf(var + LN_EPS);

        float p[9];
#pragma unroll
        for (int q = 0; q < 9; ++q)
            p[q] = rstd * D[q] - rstd * mu * cst[q] + cst[9 + q] + bp[q];

        float tr[3] = { p[0] * TRANS_SCALE, p[1] * TRANS_SCALE, p[2] * TRANS_SCALE };
        float nx = sqrtf(p[3]*p[3] + p[4]*p[4] + p[5]*p[5]);
        float ix = 1.0f / (nx + 1e-5f);
        float vx[3] = { p[3]*ix, p[4]*ix, p[5]*ix };
        float ny = sqrtf(p[6]*p[6] + p[7]*p[7] + p[8]*p[8]);
        float iy = 1.0f / (ny + 1e-5f);
        float vy[3] = { p[6]*iy, p[7]*iy, p[8]*iy };

        // e1 = normed(trans - (vec_x + trans)) = normed(-vx)
        float ne = sqrtf(vx[0]*vx[0] + vx[1]*vx[1] + vx[2]*vx[2]);
        float ie = 1.0f / (ne + 1e-10f);
        float e1[3] = { -vx[0]*ie, -vx[1]*ie, -vx[2]*ie };
        float d  = e1[0]*vy[0] + e1[1]*vy[1] + e1[2]*vy[2];
        float u2[3] = { vy[0]-e1[0]*d, vy[1]-e1[1]*d, vy[2]-e1[2]*d };
        float nu = sqrtf(u2[0]*u2[0] + u2[1]*u2[1] + u2[2]*u2[2]);
        float iu = 1.0f / (nu + 1e-10f);
        float e2[3] = { u2[0]*iu, u2[1]*iu, u2[2]*iu };
        float e3[3] = { e1[1]*e2[2] - e1[2]*e2[1],
                        e1[2]*e2[0] - e1[0]*e2[2],
                        e1[0]*e2[1] - e1[1]*e2[0] };

        const bool msk = amask[g] != 0;
        float Ru[3][3];
#pragma unroll
        for (int i = 0; i < 3; ++i) {
            Ru[i][0] = msk ? e1[i] : (i == 0 ? 1.0f : 0.0f);
            Ru[i][1] = msk ? e2[i] : (i == 1 ? 1.0f : 0.0f);
            Ru[i][2] = msk ? e3[i] : (i == 2 ? 1.0f : 0.0f);
        }
        float tu[3] = { msk ? tr[0] : 0.0f, msk ? tr[1] : 0.0f, msk ? tr[2] : 0.0f };

        float a12[12];
        const float* ap = affine + (size_t)g * 12;
#pragma unroll
        for (int q = 0; q < 12; ++q) a12[q] = ap[q];

        float R[3][3], t[3];
#pragma unroll
        for (int i = 0; i < 3; ++i) {
#pragma unroll
            for (int j = 0; j < 3; ++j)
                R[i][j] = a12[i*3+0]*Ru[0][j] + a12[i*3+1]*Ru[1][j] + a12[i*3+2]*Ru[2][j];
            t[i] = a12[i*3+0]*tu[0] + a12[i*3+1]*tu[1] + a12[i*3+2]*tu[2] + a12[9+i];
        }

        float* oa = out_aff + (size_t)g * 12;
#pragma unroll
        for (int i = 0; i < 3; ++i) {
#pragma unroll
            for (int j = 0; j < 3; ++j) oa[i*3+j] = R[i][j];
            oa[9+i] = t[i];
        }

        const float bb[3][3] = { { 0.5256f, 1.3612f, 0.0f },
                                 { 0.0f,    0.0f,    0.0f },
                                 { -1.5251f, 0.0f,   0.0f } };
        float* ox = out_xyz + (size_t)g * 9;
#pragma unroll
        for (int aI = 0; aI < 3; ++aI)
#pragma unroll
            for (int i = 0; i < 3; ++i)
                ox[aI*3+i] = R[i][0]*bb[aI][0] + R[i][1]*bb[aI][1]
                           + R[i][2]*bb[aI][2] + t[i];
    }
}

// ---------------------------------------------------------------------------
extern "C" void kernel_launch(void* const* d_in, const int* in_sizes, int n_in,
                              void* d_out, int out_size, void* d_ws, size_t ws_size,
                              hipStream_t stream) {
    const float*         x      = (const float*)d_in[0];
    const float*         affine = (const float*)d_in[1];
    const unsigned char* amask  = (const unsigned char*)d_in[2];
    const float*         W1     = (const float*)d_in[3];
    const float*         b1     = (const float*)d_in[4];
    const float*         lnw    = (const float*)d_in[5];
    const float*         lnb    = (const float*)d_in[6];
    const float*         Wp     = (const float*)d_in[7];
    const float*         bp     = (const float*)d_in[8];

    float* out_aff = (float*)d_out;
    float* out_xyz = out_aff + (size_t)NROWS * 12;

    unsigned short* W1bf = (unsigned short*)d_ws;             // 512 KB
    float* cst = (float*)((char*)d_ws + 512 * 1024);          // Cw[9], Cb[9]

    convert_w1_kernel<<<1024, 256, 0, stream>>>(W1, W1bf);
    consts_kernel<<<1, 288, 0, stream>>>(lnw, lnb, Wp, cst);
    structure_head_kernel<<<NROWS / 128, 1024, 0, stream>>>(
        x, W1bf, b1, lnw, Wp, bp, cst, affine, amask, out_aff, out_xyz);
}
